// SCNAttnDecoder_32289564131585
// MI455X (gfx1250) — matrix operand order repaired and checked by reading, not compiled, not run on hardware
//
#include <hip/hip_runtime.h>
#include <hip/hip_bf16.h>

// ---------------------------------------------------------------------------
// SCNAttnDecoder forward on MI455X (gfx1250, wave32, WMMA + TDM).
//   fp32 WMMA (16x16x4) for the factored-LSTM GEMMs, bf16 WMMA (16x16x32)
//   for the 63-GFLOP output projection (bandwidth-bound at 23.3 TB/s).
//   Projection stages its shared 32x512 bf16 A-panel into LDS via the
//   Tensor Data Mover (tensor_load_to_lds / s_wait_tensorcnt).
// ---------------------------------------------------------------------------

typedef __attribute__((ext_vector_type(2)))  float    v2f;
typedef __attribute__((ext_vector_type(4)))  float    v4f;
typedef __attribute__((ext_vector_type(8)))  float    v8f;
typedef __attribute__((ext_vector_type(8)))  __bf16   v8bf;
typedef __attribute__((ext_vector_type(16))) __bf16   v16bf;
typedef __attribute__((ext_vector_type(4)))  unsigned v4u;
typedef __attribute__((ext_vector_type(4)))  int      v4i;
typedef __attribute__((ext_vector_type(8)))  int      v8i;
typedef __attribute__((address_space(3)))    char     lds_char_t;

#define DEV static __device__ __forceinline__

DEV v8f wmma_f32(v2f a, v2f b, v8f c) {
  // D = A(16x4 f32) * B(4x16 f32) + C(16x16 f32)
  return __builtin_amdgcn_wmma_f32_16x16x4_f32(false, a, false, b, (short)0, c,
                                               false, false);
}
DEV v8f wmma_bf16(v16bf a, v16bf b, v8f c) {
  // D = A(16x32 bf16) * B(32x16 bf16) + C(16x16 f32)
  return __builtin_amdgcn_wmma_f32_16x16x32_bf16(false, a, false, b, (short)0, c,
                                                 false, false);
}

DEV float sigmoidf_(float x) { return 1.0f / (1.0f + __expf(-x)); }

// ---------------------------------------------------------------------------
// fp32 per-wave GEMM cores.  A row-major [*, lda], B row-major [K, ldb].
// A-fragment (16x4): lane m=lane&15, h=lane>>4 holds K = k0+2h+{0,1} (v2f).
// B-fragment (4x16): lane n=lane&15, h        holds K = k0+2h+{0,1}.
// C/D (16x16): vgpr e -> row m0 + e + 8h, col n0 + n.
// ---------------------------------------------------------------------------
DEV void gemm_f32_1x1(const float* __restrict__ At,  // A + m0*lda
                      int lda,
                      const float* __restrict__ Bt,  // B + n0
                      int ldb, int K, v8f& acc) {
  const int lane = threadIdx.x & 31;
  const int m = lane & 15;
  const int h = lane >> 4;
  const float* Ar = At + (size_t)m * lda;
  const float* Bc = Bt + m;  // n == lane&15
  for (int k0 = 0; k0 < K; k0 += 4) {
    const int k = k0 + 2 * h;
    v2f a = *(const v2f*)(Ar + k);
    v2f b;
    b[0] = Bc[(size_t)k * ldb];
    b[1] = Bc[(size_t)(k + 1) * ldb];
    acc = wmma_f32(a, b, acc);
  }
}

DEV void gemm_f32_2x2(const float* __restrict__ At,  // A + m0*lda (32 rows)
                      int lda,
                      const float* __restrict__ Bt,  // B + n0 (32 cols)
                      int ldb, int K, v8f acc[2][2]) {
  const int lane = threadIdx.x & 31;
  const int m = lane & 15;
  const int h = lane >> 4;
  const float* Ar0 = At + (size_t)m * lda;
  const float* Ar1 = At + (size_t)(m + 16) * lda;
  const float* Bc0 = Bt + m;
  const float* Bc1 = Bt + m + 16;
  for (int k0 = 0; k0 < K; k0 += 4) {
    const int k = k0 + 2 * h;
    v2f a0 = *(const v2f*)(Ar0 + k);
    v2f a1 = *(const v2f*)(Ar1 + k);
    v2f b0, b1;
    b0[0] = Bc0[(size_t)k * ldb];
    b0[1] = Bc0[(size_t)(k + 1) * ldb];
    b1[0] = Bc1[(size_t)k * ldb];
    b1[1] = Bc1[(size_t)(k + 1) * ldb];
    acc[0][0] = wmma_f32(a0, b0, acc[0][0]);
    acc[0][1] = wmma_f32(a0, b1, acc[0][1]);
    acc[1][0] = wmma_f32(a1, b0, acc[1][0]);
    acc[1][1] = wmma_f32(a1, b1, acc[1][1]);
  }
}

DEV v8f zero8() { v8f z = {0.f,0.f,0.f,0.f,0.f,0.f,0.f,0.f}; return z; }

// bf16 16-element fragment from two contiguous 16B chunks
DEV v16bf ld16bf(const __bf16* p0, const __bf16* p1) {
  v8bf lo = *(const v8bf*)p0;
  v8bf hi = *(const v8bf*)p1;
  v16bf r;
#pragma unroll
  for (int i = 0; i < 8; ++i) { r[i] = lo[i]; r[i + 8] = hi[i]; }
  return r;
}

// ---------------------------------------------------------------------------
// Problem constants
// ---------------------------------------------------------------------------
#define BB   128
#define TT   24
#define EE   512
#define HH   512
#define RR   512
#define NTG  400
#define VV   20000
#define MROW (TT * BB)       // 3072
#define NTIL (VV / 16)       // 1250 column tiles of output
#define KC   (HH / 16)       // 32 K-strips of 16 for packed Wout

// ---------------------------------------------------------------------------
// Kernel 1: build decoder inputs X[t*128+b][e] = (t==0 ? 0 : emb[cap[b][t-1]][e])
// ---------------------------------------------------------------------------
__global__ void k_embed(const int* __restrict__ cap,
                        const float* __restrict__ emb,
                        float* __restrict__ X) {
  int idx = blockIdx.x * blockDim.x + threadIdx.x;   // one float4 each
  if (idx >= MROW * (EE / 4)) return;
  int e4 = idx & 127;
  int r  = idx >> 7;
  int t  = r >> 7;
  int b  = r & 127;
  v4f v = {0.f, 0.f, 0.f, 0.f};
  if (t != 0) {
    int tok = cap[b * TT + (t - 1)];
    v = ((const v4f*)emb)[(size_t)tok * (EE / 4) + e4];
  }
  ((v4f*)X)[idx] = v;
}

// ---------------------------------------------------------------------------
// Kernel 2: initial state h = h0[0], c = c0[0]
// ---------------------------------------------------------------------------
__global__ void k_init(const float* __restrict__ h0, const float* __restrict__ c0,
                       float* __restrict__ h, float* __restrict__ c) {
  int idx = blockIdx.x * blockDim.x + threadIdx.x;
  if (idx < BB * HH) { h[idx] = h0[idx]; c[idx] = c0[idx]; }
}

// ---------------------------------------------------------------------------
// Kernel 3: temp2[g] = s @ Wb[g],  temp5[g] = s @ Ub[g]   (M=128,K=400,N=512)
// 512 waves: set(2) x g(4) x mb(4) x nb(16), 32x32 tile per wave.
// ---------------------------------------------------------------------------
__global__ void k_tags(const float* __restrict__ s,
                       const float* __restrict__ Wb,
                       const float* __restrict__ Ub,
                       float* __restrict__ temp2,
                       float* __restrict__ temp5) {
  int w   = blockIdx.x * (blockDim.x >> 5) + (threadIdx.x >> 5);
  int nb  = w & 15, mb = (w >> 4) & 3, g = (w >> 6) & 3, set = (w >> 8) & 1;
  int m0  = mb * 32, n0 = nb * 32;
  const float* Bmat = (set ? Ub : Wb) + (size_t)g * NTG * RR;
  float* out        = (set ? temp5 : temp2) + (size_t)g * BB * RR;
  v8f acc[2][2] = {zero8(), zero8(), zero8(), zero8()};
  gemm_f32_2x2(s + (size_t)m0 * NTG, NTG, Bmat + n0, RR, NTG, acc);
  int lane = threadIdx.x & 31, n = lane & 15, h = lane >> 4;
#pragma unroll
  for (int i = 0; i < 2; ++i)
#pragma unroll
    for (int j = 0; j < 2; ++j)
#pragma unroll
      for (int e = 0; e < 8; ++e) {
        int row = m0 + 16 * i + e + 8 * h;
        int col = n0 + 16 * j + n;
        out[(size_t)row * RR + col] = acc[i][j][e];
      }
}

// ---------------------------------------------------------------------------
// Kernel 4: T1s[g][r][:] = (X[r] @ Wa[g]) * temp2[g][r%128][:]
// 6144 waves: g(4) x mb(96) x nb(16)
// ---------------------------------------------------------------------------
__global__ void k_c1(const float* __restrict__ X,
                     const float* __restrict__ Wa,
                     const float* __restrict__ temp2,
                     float* __restrict__ T1s) {
  int w  = blockIdx.x * (blockDim.x >> 5) + (threadIdx.x >> 5);
  int g  = w / 1536;
  int rem = w % 1536;
  int mb = rem >> 4, nb = rem & 15;
  int m0 = mb * 32, n0 = nb * 32;
  v8f acc[2][2] = {zero8(), zero8(), zero8(), zero8()};
  gemm_f32_2x2(X + (size_t)m0 * EE, EE,
               Wa + (size_t)g * EE * RR + n0, RR, EE, acc);
  int lane = threadIdx.x & 31, n = lane & 15, h = lane >> 4;
#pragma unroll
  for (int i = 0; i < 2; ++i)
#pragma unroll
    for (int j = 0; j < 2; ++j)
#pragma unroll
      for (int e = 0; e < 8; ++e) {
        int row = m0 + 16 * i + e + 8 * h;
        int col = n0 + 16 * j + n;
        float scale = temp2[((size_t)g * BB + (row & 127)) * RR + col];
        T1s[((size_t)g * MROW + row) * RR + col] = acc[i][j][e] * scale;
      }
}

// ---------------------------------------------------------------------------
// Kernel 5: GX[t][g][b][:] = T1s[g][t*128+b] @ Wc[g] + bias[g]
// ---------------------------------------------------------------------------
__global__ void k_c2(const float* __restrict__ T1s,
                     const float* __restrict__ Wc,
                     const float* __restrict__ bias,
                     float* __restrict__ GX) {
  int w  = blockIdx.x * (blockDim.x >> 5) + (threadIdx.x >> 5);
  int g  = w / 1536;
  int rem = w % 1536;
  int mb = rem >> 4, nb = rem & 15;
  int m0 = mb * 32, n0 = nb * 32;
  v8f acc[2][2] = {zero8(), zero8(), zero8(), zero8()};
  gemm_f32_2x2(T1s + ((size_t)g * MROW + m0) * RR, RR,
               Wc + (size_t)g * RR * HH + n0, HH, RR, acc);
  int lane = threadIdx.x & 31, n = lane & 15, h = lane >> 4;
#pragma unroll
  for (int i = 0; i < 2; ++i)
#pragma unroll
    for (int j = 0; j < 2; ++j)
#pragma unroll
      for (int e = 0; e < 8; ++e) {
        int row = m0 + 16 * i + e + 8 * h;     // t*128 + b
        int col = n0 + 16 * j + n;
        int t = row >> 7, b = row & 127;
        GX[(((size_t)t * 4 + g) * BB + b) * HH + col] =
            acc[i][j][e] + bias[g * HH + col];
      }
}

// ---------------------------------------------------------------------------
// Kernel 6: pack Wout[H][V] (f32) into bf16 B-fragment order:
//   Bp[((nt*32 + kc)*16 + n)*16 + j] = bf16(Wout[(kc*16+j)*V + nt*16 + n])
// so a lane's 16-K strip (K = kc*16 .. +15) is 32 contiguous bytes.
// ---------------------------------------------------------------------------
__global__ void k_pack_wout(const float* __restrict__ Wout,
                            __bf16* __restrict__ Bp) {
  int idx = blockIdx.x * blockDim.x + threadIdx.x;
  if (idx >= NTIL * KC * 16 * 16) return;
  int j  = idx & 15;
  int n  = (idx >> 4) & 15;
  int kc = (idx >> 8) & 31;
  int nt = idx >> 13;
  Bp[idx] = (__bf16)Wout[(size_t)(kc * 16 + j) * VV + nt * 16 + n];
}

// ---------------------------------------------------------------------------
// Kernel 7 (per step): T6s[g] = (h @ Ua[g]) * temp5[g]
// 256 waves: g(4) x mb(4) x nb(16)
// ---------------------------------------------------------------------------
__global__ void k_t6(const float* __restrict__ h,
                     const float* __restrict__ Ua,
                     const float* __restrict__ temp5,
                     float* __restrict__ T6s) {
  int w  = blockIdx.x * (blockDim.x >> 5) + (threadIdx.x >> 5);
  int g  = (w >> 6) & 3;
  int mb = (w >> 4) & 3, nb = w & 15;
  int m0 = mb * 32, n0 = nb * 32;
  v8f acc[2][2] = {zero8(), zero8(), zero8(), zero8()};
  gemm_f32_2x2(h + (size_t)m0 * HH, HH,
               Ua + (size_t)g * HH * RR + n0, RR, HH, acc);
  int lane = threadIdx.x & 31, n = lane & 15, hh = lane >> 4;
#pragma unroll
  for (int i = 0; i < 2; ++i)
#pragma unroll
    for (int j = 0; j < 2; ++j)
#pragma unroll
      for (int e = 0; e < 8; ++e) {
        int row = m0 + 16 * i + e + 8 * hh;
        int col = n0 + 16 * j + n;
        size_t idx = ((size_t)g * BB + row) * RR + col;
        T6s[idx] = acc[i][j][e] * temp5[idx];
      }
}

// ---------------------------------------------------------------------------
// Kernel 8 (per step): wave computes its 16x16 tile of ALL 4 gates of
//   gh = T6s[g] @ Uc[g], then pre = gh + GX[t][g], applies LSTM update
//   in place (c,h), and emits hs row as bf16 for the projection.
// 256 waves: mb(8) x nb(32)
// ---------------------------------------------------------------------------
__global__ void k_gh_upd(const float* __restrict__ T6s,
                         const float* __restrict__ Uc,
                         const float* __restrict__ GX,
                         float* __restrict__ c,
                         float* __restrict__ h,
                         __bf16* __restrict__ Hsb,
                         int t) {
  int w  = blockIdx.x * (blockDim.x >> 5) + (threadIdx.x >> 5);
  int mb = w >> 5, nb = w & 31;
  int m0 = mb * 16, n0 = nb * 16;
  v8f acc[4];
#pragma unroll
  for (int g = 0; g < 4; ++g) {
    acc[g] = zero8();
    gemm_f32_1x1(T6s + ((size_t)g * BB + m0) * RR, RR,
                 Uc + (size_t)g * RR * HH + n0, HH, RR, acc[g]);
  }
  const float* GXt = GX + (size_t)t * 4 * BB * HH;
  int lane = threadIdx.x & 31, n = lane & 15, hh = lane >> 4;
#pragma unroll
  for (int e = 0; e < 8; ++e) {
    int row = m0 + e + 8 * hh;     // b
    int col = n0 + n;
    size_t gofs = (size_t)row * HH + col;
    float pi = acc[0][e] + GXt[(size_t)(0 * BB) * HH + gofs];
    float pf = acc[1][e] + GXt[(size_t)(1 * BB) * HH + gofs];
    float po = acc[2][e] + GXt[(size_t)(2 * BB) * HH + gofs];
    float pc = acc[3][e] + GXt[(size_t)(3 * BB) * HH + gofs];
    float ig = sigmoidf_(pi);
    float fg = sigmoidf_(pf);
    float og = sigmoidf_(po);
    float ct = tanhf(pc);
    float cn = fg * c[gofs] + ig * ct;
    c[gofs] = cn;
    float hn = og * tanhf(cn);
    h[gofs] = hn;
    Hsb[((size_t)t * BB + row) * HH + col] = (__bf16)hn;
  }
}

// ---------------------------------------------------------------------------
// Kernel 9: logits = hs @ Wout + bout, written as out[b][t][v].
// bf16 WMMA 16x16x32. Block = 32 rows x 512 cols; the shared 32x512 bf16
// A-panel (32 KB) is DMA'd into LDS by the Tensor Data Mover once, then all
// 8 waves read A-fragments from LDS while streaming B from global.
// Grid: rb(96) x cg(40); wave wv handles col-tiles cg*32 + wv*4 + {0..3}.
// ---------------------------------------------------------------------------
__global__ void k_proj(const __bf16* __restrict__ Hsb,
                       const __bf16* __restrict__ Bp,
                       const float* __restrict__ bout,
                       float* __restrict__ out) {
  extern __shared__ __align__(16) char smem[];           // 32*512 bf16 = 32 KB
  __bf16* As = (__bf16*)smem;

  int cg = blockIdx.x % 40;
  int rb = blockIdx.x / 40;
  int m0 = rb * 32;

  // --- TDM: stage A-panel rows [m0, m0+32) of Hsb into LDS (wave 0 only) ---
  if (threadIdx.x < 32) {
    unsigned lds_base = (unsigned)(unsigned long long)(lds_char_t*)smem;
    unsigned long long ga =
        (unsigned long long)(const char*)(Hsb + (size_t)m0 * HH);
    v4u g0;
    g0[0] = 1u;                                   // count=1, user descriptor
    g0[1] = lds_base;                             // lds_addr (bytes)
    g0[2] = (unsigned)ga;                         // global_addr[31:0]
    g0[3] = (unsigned)((ga >> 32) & 0x1ffffffu)   // global_addr[56:32]
            | (2u << 30);                         // type = 2 ("image")
    v8i g1;
    g1[0] = (int)(1u << 16);                      // data_size = 1 (2 bytes)
    g1[1] = (int)(512u << 16);                    // tensor_dim0 = 512
    g1[2] = (int)(32u << 16);                     // tensor_dim1 = 32
    g1[3] = (int)(512u << 16);                    // tile_dim0 = 512
    g1[4] = 32;                                   // tile_dim1 = 32
    g1[5] = 512;                                  // tensor_dim0_stride = 512
    g1[6] = 0;
    g1[7] = 0;
    v4i gz4 = {0, 0, 0, 0};                       // groups 2/3 unused (2-D)
    v8i gz8 = {0, 0, 0, 0, 0, 0, 0, 0};
    __builtin_amdgcn_tensor_load_to_lds(g0, g1, gz4, gz4, gz8, 0);
    __builtin_amdgcn_s_wait_tensorcnt(0);
  }
  __syncthreads();

  int wv  = threadIdx.x >> 5;
  int nt0 = cg * 32 + wv * 4;
  int jmax = NTIL - nt0;
  if (jmax > 4) jmax = 4;

  int lane = threadIdx.x & 31, n = lane & 15, hh = lane >> 4;

  if (jmax > 0) {
    v8f acc[2][4];
#pragma unroll
    for (int i = 0; i < 2; ++i)
#pragma unroll
      for (int j = 0; j < 4; ++j) acc[i][j] = zero8();

    const __bf16* Ar0 = As + (size_t)n * HH;        // local row n   (tile 0)
    const __bf16* Ar1 = As + (size_t)(16 + n) * HH; // local row 16+n (tile 1)

    for (int k0 = 0; k0 < HH; k0 += 32) {
      v16bf a0 = ld16bf(Ar0 + k0 + 16 * hh, Ar0 + k0 + 16 * hh + 8);
      v16bf a1 = ld16bf(Ar1 + k0 + 16 * hh, Ar1 + k0 + 16 * hh + 8);
      int kc = (k0 >> 4) + hh;
#pragma unroll
      for (int j = 0; j < 4; ++j) {
        if (j >= jmax) break;
        const __bf16* bp = Bp + (((size_t)(nt0 + j) * KC + kc) * 16 + n) * 16;
        v16bf bf = ld16bf(bp, bp + 8);
        acc[0][j] = wmma_bf16(a0, bf, acc[0][j]);
        acc[1][j] = wmma_bf16(a1, bf, acc[1][j]);
      }
    }

#pragma unroll
    for (int i = 0; i < 2; ++i)
#pragma unroll
      for (int j = 0; j < 4; ++j) {
        if (j >= jmax) break;
#pragma unroll
        for (int e = 0; e < 8; ++e) {
          int row = m0 + 16 * i + e + 8 * hh;   // t*128 + b
          int col = (nt0 + j) * 16 + n;
          int t = row >> 7, b = row & 127;
          out[((size_t)b * TT + t) * VV + col] = acc[i][j][e] + bout[col];
        }
      }
  }
}

// ---------------------------------------------------------------------------
// Host launcher
// ---------------------------------------------------------------------------
extern "C" void kernel_launch(void* const* d_in, const int* in_sizes, int n_in,
                              void* d_out, int out_size, void* d_ws, size_t ws_size,
                              hipStream_t stream) {
  const int*   captions = (const int*)d_in[0];
  const float* s        = (const float*)d_in[1];
  const float* h0       = (const float*)d_in[2];
  const float* c0       = (const float*)d_in[3];
  const float* Wa       = (const float*)d_in[4];
  const float* Wb       = (const float*)d_in[5];
  const float* Wc       = (const float*)d_in[6];
  const float* Ua       = (const float*)d_in[7];
  const float* Ub       = (const float*)d_in[8];
  const float* Uc       = (const float*)d_in[9];
  const float* bias     = (const float*)d_in[10];
  const float* emb      = (const float*)d_in[11];
  const float* Wout     = (const float*)d_in[12];
  const float* bout     = (const float*)d_in[13];
  float* out            = (float*)d_out;

  // workspace layout (floats), with lifetime overlays
  float* ws    = (float*)d_ws;
  float* temp2 = ws;                       // 4*128*512          =   262144
  float* temp5 = temp2 + 262144;           //                     +  262144
  float* GX    = temp5 + 262144;           // 24*4*128*512       = 6291456
  float* T6s   = GX + 6291456;             // 4*128*512          =  262144
  float* hbuf  = T6s + 262144;             // 128*512            =   65536
  float* cbuf  = hbuf + 65536;             // 128*512            =   65536
  float* X     = cbuf + 65536;             // 3072*512           = 1572864
  __bf16* Hsb  = (__bf16*)X;               // overlay: 3072*512 bf16 (X dead)
  float* T1s   = X + 1572864;              // 4*3072*512         = 6291456
  __bf16* Bp   = (__bf16*)T1s;             // overlay: 512*20000 bf16 (T1s dead)

  const dim3 blk(256);

  // setup
  k_embed<<<dim3(1536), blk, 0, stream>>>(captions, emb, X);
  k_init <<<dim3(256),  blk, 0, stream>>>(h0, c0, hbuf, cbuf);
  k_tags <<<dim3(64),   blk, 0, stream>>>(s, Wb, Ub, temp2, temp5);
  k_c1   <<<dim3(768),  blk, 0, stream>>>(X, Wa, temp2, T1s);
  k_c2   <<<dim3(768),  blk, 0, stream>>>(T1s, Wc, bias, GX);
  k_pack_wout<<<dim3(40000), blk, 0, stream>>>(Wout, Bp);  // T1s dead now

  // recurrence: 24 sequential steps, 2 kernels each (kernel boundary = sync)
  for (int t = 0; t < TT; ++t) {
    k_t6    <<<dim3(32), blk, 0, stream>>>(hbuf, Ua, temp5, T6s);
    k_gh_upd<<<dim3(32), blk, 0, stream>>>(T6s, Uc, GX, cbuf, hbuf, Hsb, t);
  }

  // output projection (bf16 WMMA + TDM-staged LDS A-panel)
  k_proj<<<dim3(96 * 40), blk, 32768, stream>>>(Hsb, Bp, bout, out);
}
